// TemporalAttentionDecoder_9543417332542
// MI455X (gfx1250) — matrix operand order repaired and checked by reading, not compiled, not run on hardware
//
#include <hip/hip_runtime.h>
#include <stdint.h>

// ---------------------------------------------------------------------------
// Problem constants (from reference): B=2048, T=64, M=256, P=256
// ---------------------------------------------------------------------------
#define BB 2048
#define TT 64
#define MM 256
#define PP 256

typedef __attribute__((ext_vector_type(16))) __bf16 v16bf;
typedef __attribute__((ext_vector_type(8)))  __bf16 v8bf;
typedef __attribute__((ext_vector_type(8)))  float  v8f;

// gfx1250 async global->LDS path (ASYNCcnt-tracked). The async-load itself is
// issued via inline asm (the clang builtin's pointer types differ between
// toolchains); the counter wait uses the probe-confirmed builtin.
#if defined(__AMDGCN__) && __has_builtin(__builtin_amdgcn_s_wait_asynccnt)
#define HAVE_ASYNC_LDS 1
#else
#define HAVE_ASYNC_LDS 0
#endif

__device__ __forceinline__ unsigned short f2bf(float f) {
    unsigned u = __float_as_uint(f);
    u += 0x7FFFu + ((u >> 16) & 1u);          // round-to-nearest-even
    return (unsigned short)(u >> 16);
}
__device__ __forceinline__ float bf2f(unsigned short h) {
    return __uint_as_float((unsigned)h << 16);
}
__device__ __forceinline__ float sigf(float x) { return 1.f / (1.f + __expf(-x)); }

// A-matrix fragment, 16-bit 16x32 (ISA 7.12.2): lane half h holds two 8-elem
// contiguous K chunks at k = h*8..h*8+7 and 16+h*8..16+h*8+7.
__device__ __forceinline__ v16bf load_fragA(const unsigned short* base, int hf) {
    union { v16bf v; v8bf h[2]; } u;
    u.h[0] = *(const v8bf*)(base + hf * 8);
    u.h[1] = *(const v8bf*)(base + 16 + hf * 8);
    return u.v;
}
// B-matrix fragment (16-bit): lane half h holds two 8-elem K chunks at
// k = h*8..h*8+7 and 16+h*8..16+h*8+7 (same order as A).
__device__ __forceinline__ v16bf load_fragB(const unsigned short* base, int hf) {
    union { v16bf v; v8bf h[2]; } u;
    u.h[0] = *(const v8bf*)(base + hf * 8);
    u.h[1] = *(const v8bf*)(base + 16 + hf * 8);
    return u.v;
}

// ---------------------------------------------------------------------------
// Generic bf16 WMMA GEMM:  C[Mrows,N] = A[Mrows,K] * Bw[N,K]^T (+bias[N])
// One wave -> 16x64 tile; K is a compile-time constant so the software-
// pipelined K loop fully unrolls (cross-iteration fragment copies become
// SSA renames, loads spread across the body, partial loadcnt waits).
// Requires Mrows%16==0, N%64==0, K%32==0.
// ---------------------------------------------------------------------------
template <int K, bool OUT_BF, bool ADD_BIAS>
__global__ void gemm_bf16_wmma(const unsigned short* __restrict__ A,
                               const unsigned short* __restrict__ Bw,
                               const float* __restrict__ bias,
                               void* __restrict__ Cout,
                               int Mrows, int N) {
    const int wavesPerBlock = blockDim.x >> 5;
    int wv   = blockIdx.x * wavesPerBlock + (threadIdx.x >> 5);
    int lane = threadIdx.x & 31;
    int tilesN = N >> 6;
    int tm = wv / tilesN;
    int tn = wv - tm * tilesN;
    if (tm * 16 >= Mrows) return;
    int row0 = tm << 4, col0 = tn << 6;
    int hf = lane >> 4, l16 = lane & 15;

    const unsigned short* Arow = A  + (size_t)(row0 + l16) * K;
    const unsigned short* Brow = Bw + (size_t)(col0 + l16) * K;

    v8f z;
#pragma unroll
    for (int r = 0; r < 8; ++r) z[r] = 0.f;
    v8f acc[4] = {z, z, z, z};

    // prologue: fragments for kb = 0
    v16bf a_cur = load_fragA(Arow, hf);
    v16bf b_cur[4];
#pragma unroll
    for (int j = 0; j < 4; ++j)
        b_cur[j] = load_fragB(Brow + (size_t)(j * 16) * K, hf);

#pragma unroll
    for (int kb = 0; kb < K; kb += 32) {
        v16bf a_nxt = a_cur;
        v16bf b_nxt[4] = {b_cur[0], b_cur[1], b_cur[2], b_cur[3]};
        const int kn = kb + 32;
        if (kn < K) {                       // prefetch next k-block
            a_nxt = load_fragA(Arow + kn, hf);
#pragma unroll
            for (int j = 0; j < 4; ++j)
                b_nxt[j] = load_fragB(Brow + (size_t)(j * 16) * K + kn, hf);
        }
#pragma unroll
        for (int j = 0; j < 4; ++j)
            acc[j] = __builtin_amdgcn_wmma_f32_16x16x32_bf16(
                false, a_cur, false, b_cur[j], (short)0, acc[j], false, false);
        a_cur = a_nxt;
#pragma unroll
        for (int j = 0; j < 4; ++j) b_cur[j] = b_nxt[j];
    }

    // C/D layout: lane L -> col = col0 + L%16, rows m = row0 + (L/16)*8 + r
    int mb = row0 + hf * 8;
#pragma unroll
    for (int j = 0; j < 4; ++j) {
        int col = col0 + j * 16 + l16;
        float bv = ADD_BIAS ? bias[col] : 0.f;
#pragma unroll
        for (int r = 0; r < 8; ++r) {
            float v = acc[j][r] + bv;
            size_t idx = (size_t)(mb + r) * N + col;
            if (OUT_BF) ((unsigned short*)Cout)[idx] = f2bf(v);
            else        ((float*)Cout)[idx] = v;
        }
    }
}

// ---------------------------------------------------------------------------
// Small helper kernels
// ---------------------------------------------------------------------------
__global__ void f32_to_bf16_k(const float* __restrict__ src,
                              unsigned short* __restrict__ dst, size_t n) {
    size_t i = (size_t)blockIdx.x * blockDim.x + threadIdx.x;
    size_t stride = (size_t)gridDim.x * blockDim.x;
    for (; i < n; i += stride) dst[i] = f2bf(src[i]);
}

__global__ void fill_zero_k(float* __restrict__ p, size_t n) {
    size_t i = (size_t)blockIdx.x * blockDim.x + threadIdx.x;
    size_t stride = (size_t)gridDim.x * blockDim.x;
    for (; i < n; i += stride) p[i] = 0.f;
}

// ds16[b, 0:256] = bf16(d[b,:]); ds16[b, 256:512] = bf16(s[b,:]); d16 = bf16(d)
__global__ void pack_ds_k(const float* __restrict__ d, const float* __restrict__ s,
                          unsigned short* __restrict__ ds16,
                          unsigned short* __restrict__ d16) {
    size_t i = (size_t)blockIdx.x * blockDim.x + threadIdx.x;   // over B*P
    int b = (int)(i >> 8), p = (int)(i & 255);
    unsigned short dv = f2bf(d[i]);
    ds16[(size_t)b * 512 + p]       = dv;
    ds16[(size_t)b * 512 + 256 + p] = f2bf(s[i]);
    if (d16) d16[i] = dv;
}

// ---------------------------------------------------------------------------
// Fused attention step (one block = one batch row, 256 threads = 8 waves):
//   l_t[t] = sum_m tanh(x1[b,m] + Ue[b,t,m]) * v_d[m]
//   beta   = softmax(l_t); c_t[b,m] = sum_t beta[t] * enc[b,t,m]
// Pass 1 stages Ue rows LDS-side with GLOBAL_LOAD_ASYNC_TO_LDS_B128
// (one row = 32 lanes x 16B), double-buffered per wave, ASYNCcnt-waited.
// ---------------------------------------------------------------------------
__global__ void attention_step_k(const float* __restrict__ x1,
                                 const unsigned short* __restrict__ Ue16,
                                 const unsigned short* __restrict__ E16,
                                 const float* __restrict__ v_d,
                                 float* __restrict__ c_t,
                                 float* __restrict__ beta_out) {
    __shared__ float sx[MM], sv[MM], sl[TT], sinv;
    int b = blockIdx.x, tid = threadIdx.x;
    sx[tid] = x1[(size_t)b * MM + tid];
    sv[tid] = v_d[tid];
    __syncthreads();

    int wv = tid >> 5, lane = tid & 31;

#if HAVE_ASYNC_LDS
    __shared__ __align__(16) unsigned short stage[8][2][MM];   // 8 KB
    {
        const unsigned short* base = Ue16 + (size_t)b * TT * MM;
        auto issue = [&](int t, int buf) {
            // per-lane 16B: lane copies bytes [lane*16, lane*16+16) of row t
            unsigned long long gaddr =
                (unsigned long long)(uintptr_t)(base + (size_t)t * MM + lane * 8);
            unsigned lds_off =
                (unsigned)(uintptr_t)(&stage[wv][buf][lane * 8]);
            asm volatile("global_load_async_to_lds_b128 %0, %1, off"
                         :: "v"(lds_off), "v"(gaddr)
                         : "memory");
        };
        issue(wv, 0);
        int i = 0;
        for (int t = wv; t < TT; t += 8, ++i) {
            int cur = i & 1;
            if (t + 8 < TT) {
                issue(t + 8, cur ^ 1);                  // prefetch next row
                __builtin_amdgcn_s_wait_asynccnt(1);    // current row landed
            } else {
                __builtin_amdgcn_s_wait_asynccnt(0);
            }
            float acc = 0.f;
#pragma unroll
            for (int j = 0; j < 8; ++j) {
                int m = lane + j * 32;
                acc += tanhf(sx[m] + bf2f(stage[wv][cur][m])) * sv[m];
            }
#pragma unroll
            for (int off = 16; off > 0; off >>= 1) acc += __shfl_down(acc, off, 32);
            if (lane == 0) sl[t] = acc;
        }
    }
#else
    for (int t = wv; t < TT; t += 8) {
        const unsigned short* row = Ue16 + ((size_t)b * TT + t) * MM;
        float acc = 0.f;
#pragma unroll
        for (int j = 0; j < 8; ++j) {
            int m = lane + j * 32;
            acc += tanhf(sx[m] + bf2f(row[m])) * sv[m];
        }
#pragma unroll
        for (int off = 16; off > 0; off >>= 1) acc += __shfl_down(acc, off, 32);
        if (lane == 0) sl[t] = acc;
    }
#endif
    __syncthreads();

    if (tid == 0) {
        float mx = sl[0];
        for (int t = 1; t < TT; ++t) mx = fmaxf(mx, sl[t]);
        float sum = 0.f;
        for (int t = 0; t < TT; ++t) { float e = __expf(sl[t] - mx); sl[t] = e; sum += e; }
        sinv = 1.f / sum;
    }
    __syncthreads();

    if (tid < TT) {
        float bta = sl[tid] * sinv;
        sl[tid] = bta;
        beta_out[(size_t)b * TT + tid] = bta;   // last step's values = scan carry
    }
    __syncthreads();

    const unsigned short* eb = E16 + (size_t)b * TT * MM;
    float acc = 0.f;
    for (int t = 0; t < TT; ++t) {
        if (t + 8 < TT) __builtin_prefetch(&eb[(size_t)(t + 8) * MM + tid], 0, 1);
        acc += sl[t] * bf2f(eb[(size_t)t * MM + tid]);
    }
    c_t[(size_t)b * MM + tid] = acc;
}

// y_tilda[b] = c_t[b,:]·w[0:256] + y[b,t]*w[256] + wb   (one wave per b)
__global__ void ytilda_k(const float* __restrict__ c_t, const float* __restrict__ y,
                         int t_step, const float* __restrict__ wt_w,
                         const float* __restrict__ wt_b, float* __restrict__ yt) {
    int b    = blockIdx.x * (blockDim.x >> 5) + (threadIdx.x >> 5);
    int lane = threadIdx.x & 31;
    float acc = 0.f;
#pragma unroll
    for (int j = 0; j < 8; ++j) {
        int m = lane + j * 32;
        acc += c_t[(size_t)b * MM + m] * wt_w[m];
    }
#pragma unroll
    for (int off = 16; off > 0; off >>= 1) acc += __shfl_down(acc, off, 32);
    if (lane == 0) yt[b] = acc + y[(size_t)b * TT + t_step] * wt_w[MM] + wt_b[0];
}

// LSTM pointwise:  G = d @ W_hh^T precomputed; add x*W_ih + b_ih + b_hh,
// gate order i,f,g,o;  s=c (cell), d=h (hidden)
__global__ void lstm_step_k(const float* __restrict__ G, const float* __restrict__ yt,
                            const float* __restrict__ W_ih, const float* __restrict__ b_ih,
                            const float* __restrict__ b_hh,
                            float* __restrict__ d, float* __restrict__ s) {
    size_t i = (size_t)blockIdx.x * blockDim.x + threadIdx.x;   // over B*P
    int b = (int)(i >> 8), p = (int)(i & 255);
    float x = yt[b];
    const float* g = G + (size_t)b * (4 * PP);
    float gi = g[p]            + x * W_ih[p]            + b_ih[p]            + b_hh[p];
    float gf = g[PP + p]       + x * W_ih[PP + p]       + b_ih[PP + p]       + b_hh[PP + p];
    float gg = g[2 * PP + p]   + x * W_ih[2 * PP + p]   + b_ih[2 * PP + p]   + b_hh[2 * PP + p];
    float go = g[3 * PP + p]   + x * W_ih[3 * PP + p]   + b_ih[3 * PP + p]   + b_hh[3 * PP + p];
    float cn = sigf(gf) * s[i] + sigf(gi) * tanhf(gg);
    float hn = sigf(go) * tanhf(cn);
    s[i] = cn;
    d[i] = hn;
}

// out[b] = X[b,:]·w + bias  (one wave per b)
__global__ void rowdot_k(const float* __restrict__ X, const float* __restrict__ w,
                         const float* __restrict__ bias, float* __restrict__ out) {
    int b    = blockIdx.x * (blockDim.x >> 5) + (threadIdx.x >> 5);
    int lane = threadIdx.x & 31;
    float acc = 0.f;
#pragma unroll
    for (int j = 0; j < 8; ++j) {
        int m = lane + j * 32;
        acc += X[(size_t)b * MM + m] * w[m];
    }
#pragma unroll
    for (int off = 16; off > 0; off >>= 1) acc += __shfl_down(acc, off, 32);
    if (lane == 0) out[b] = acc + bias[0];
}

// ---------------------------------------------------------------------------
// Host orchestration
// ---------------------------------------------------------------------------
extern "C" void kernel_launch(void* const* d_in, const int* in_sizes, int n_in,
                              void* d_out, int out_size, void* d_ws, size_t ws_size,
                              hipStream_t stream) {
    (void)in_sizes; (void)n_in; (void)out_size; (void)ws_size;

    const float* enc   = (const float*)d_in[0];
    const float* y     = (const float*)d_in[1];
    const float* W_d_w = (const float*)d_in[2];
    const float* W_d_b = (const float*)d_in[3];
    const float* U_d_w = (const float*)d_in[4];
    const float* v_d_w = (const float*)d_in[5];
    const float* wt_w  = (const float*)d_in[6];
    const float* wt_b  = (const float*)d_in[7];
    const float* W_y_w = (const float*)d_in[8];
    const float* W_y_b = (const float*)d_in[9];
    const float* v_y_w = (const float*)d_in[10];
    const float* v_y_b = (const float*)d_in[11];
    const float* W_ih  = (const float*)d_in[12];
    const float* W_hh  = (const float*)d_in[13];
    const float* b_ih  = (const float*)d_in[14];
    const float* b_hh  = (const float*)d_in[15];

    float* out_y    = (float*)d_out;            // [B,1]
    float* out_beta = (float*)d_out + BB;       // [B,T,1]

    // Workspace carve-out (256B aligned slabs)
    char* wp = (char*)d_ws;
    auto alloc = [&](size_t bytes) -> char* {
        char* p = wp;
        wp += (bytes + 255) & ~(size_t)255;
        return p;
    };
    unsigned short* E16   = (unsigned short*)alloc((size_t)BB * TT * MM * 2); // 64 MB
    unsigned short* Ue16  = (unsigned short*)alloc((size_t)BB * TT * MM * 2); // 64 MB
    unsigned short* Wd16  = (unsigned short*)alloc((size_t)MM * 2 * PP * 2);
    unsigned short* Ud16  = (unsigned short*)alloc((size_t)MM * MM * 2);
    unsigned short* Whh16 = (unsigned short*)alloc((size_t)4 * PP * PP * 2);
    unsigned short* Wy16  = (unsigned short*)alloc((size_t)PP * (PP + MM) * 2);
    float*          dbuf  = (float*)alloc((size_t)BB * PP * 4);
    float*          sbuf  = (float*)alloc((size_t)BB * PP * 4);
    unsigned short* ds16  = (unsigned short*)alloc((size_t)BB * 2 * PP * 2);
    unsigned short* d16   = (unsigned short*)alloc((size_t)BB * PP * 2);
    float*          x1    = (float*)alloc((size_t)BB * MM * 4);   // reused for W_y tmp
    float*          ct    = (float*)alloc((size_t)BB * MM * 4);
    float*          yt    = (float*)alloc((size_t)BB * 4);
    float*          gates = (float*)alloc((size_t)BB * 4 * PP * 4);

    // ---- one-time setup -------------------------------------------------
    f32_to_bf16_k<<<2048, 256, 0, stream>>>(enc, E16, (size_t)BB * TT * MM);
    f32_to_bf16_k<<<64, 256, 0, stream>>>(W_d_w, Wd16, (size_t)MM * 2 * PP);
    f32_to_bf16_k<<<64, 256, 0, stream>>>(U_d_w, Ud16, (size_t)MM * MM);
    f32_to_bf16_k<<<64, 256, 0, stream>>>(W_hh, Whh16, (size_t)4 * PP * PP);
    f32_to_bf16_k<<<64, 256, 0, stream>>>(W_y_w, Wy16, (size_t)PP * (PP + MM));
    fill_zero_k<<<512, 256, 0, stream>>>(dbuf, (size_t)BB * PP);
    fill_zero_k<<<512, 256, 0, stream>>>(sbuf, (size_t)BB * PP);

    // Ue[b*T+t, n] = sum_m enc[b,t,m] * U_d[n,m]   (bf16 out, fits in L2)
    {
        int waves = (BB * TT / 16) * (MM / 64);            // 32768 waves
        gemm_bf16_wmma<MM, true, false><<<waves / 8, 256, 0, stream>>>(
            E16, Ud16, nullptr, Ue16, BB * TT, MM);
    }

    // ---- 64 serially dependent decoder steps ----------------------------
    for (int t = 0; t < TT; ++t) {
        pack_ds_k<<<BB * PP / 256, 256, 0, stream>>>(dbuf, sbuf, ds16, d16);
        // x1 = ds @ W_d^T + b   [2048,512]x[512,256]
        gemm_bf16_wmma<2 * PP, false, true><<<(BB / 16) * (MM / 64) / 8, 256, 0, stream>>>(
            ds16, Wd16, W_d_b, x1, BB, MM);
        // gates = d @ W_hh^T    [2048,256]x[256,1024]  (independent of attn)
        gemm_bf16_wmma<PP, false, false><<<(BB / 16) * (4 * PP / 64) / 8, 256, 0, stream>>>(
            d16, Whh16, nullptr, gates, BB, 4 * PP);
        // attention: tanh + dot + softmax + weighted sum of encoder states
        attention_step_k<<<BB, 256, 0, stream>>>(x1, Ue16, E16, v_d_w, ct, out_beta);
        // y_tilda
        ytilda_k<<<BB / 8, 256, 0, stream>>>(ct, y, t, wt_w, wt_b, yt);
        // LSTM pointwise update of (d, s)
        lstm_step_k<<<BB * PP / 256, 256, 0, stream>>>(gates, yt, W_ih, b_ih, b_hh,
                                                       dbuf, sbuf);
    }

    // ---- head: y_Tp1 = (concat(d, c_t) @ W_y^T + b_y) @ v_y^T + b -------
    pack_ds_k<<<BB * PP / 256, 256, 0, stream>>>(dbuf, ct, ds16, nullptr);
    gemm_bf16_wmma<PP + MM, false, true><<<(BB / 16) * (PP / 64) / 8, 256, 0, stream>>>(
        ds16, Wy16, W_y_b, x1, BB, PP);
    rowdot_k<<<BB / 8, 256, 0, stream>>>(x1, v_y_w, v_y_b, out_y);
}
